// MCPActor_59863254172190
// MI455X (gfx1250) — matrix operand order repaired and checked by reading, not compiled, not run on hardware
//
#include <hip/hip_runtime.h>
#include <hip/hip_bf16.h>

// ---- problem constants ----
#define B_   1024
#define OBS_ 256
#define S_   8
#define C_   16
#define F_   1024
#define A_   32
#define OSD_ 384   // OBS + S*C

typedef __bf16 bf16;
typedef __attribute__((ext_vector_type(16))) __bf16 v16bf;
typedef __attribute__((ext_vector_type(8)))  float  v8f;

// ---------------------------------------------------------------------------
// CDNA5 async copy: global -> LDS, 16 bytes per lane, tracked by ASYNCcnt.
//   ISA 15.18.3 GLOBAL_LOAD_ASYNC_TO_LDS_B128, GV mode:
//   lds[VDST] = mem[VADDR]
// ---------------------------------------------------------------------------
__device__ __forceinline__ void async_copy_b128(unsigned ldsOff, const void* gaddr) {
    asm volatile("global_load_async_to_lds_b128 %0, %1, off"
                 :: "v"(ldsOff), "v"((unsigned long long)(size_t)gaddr)
                 : "memory");
}
__device__ __forceinline__ void wait_async0() {
    asm volatile("s_wait_asynccnt 0x0" ::: "memory");
}

// ---------------------------------------------------------------------------
// f32 -> bf16 conversion (row-major, activations)
// ---------------------------------------------------------------------------
__global__ void cvt_f32_bf16(const float* __restrict__ src, bf16* __restrict__ dst, long n) {
    long i = (long)blockIdx.x * blockDim.x + threadIdx.x;
    if (i < n) dst[i] = (bf16)src[i];
}

// ---------------------------------------------------------------------------
// f32 [c][K][N] -> bf16 transposed [c][N][K]  (weights, one-time)
// ---------------------------------------------------------------------------
__global__ void cvt_transpose_bf16(const float* __restrict__ src, bf16* __restrict__ dst,
                                   int K, int N, long total) {
    long idx = (long)blockIdx.x * blockDim.x + threadIdx.x;
    if (idx >= total) return;
    long kn = (long)K * N;
    long c  = idx / kn;
    long r  = idx - c * kn;
    int  n  = (int)(r / K);
    int  k  = (int)(r - (long)n * K);
    dst[idx] = (bf16)src[c * kn + (long)k * N + n];
}

// ---------------------------------------------------------------------------
// skill_ft[b,c,f] = sum_s skills[b,c,s] * se_W[c,s,f] + se_b[c,f]   (K=8, VALU)
// ---------------------------------------------------------------------------
__global__ void skill_ft_kernel(const float* __restrict__ obs_skill,
                                const float* __restrict__ seW,
                                const float* __restrict__ seB,
                                bf16* __restrict__ out) {
    long idx = (long)blockIdx.x * blockDim.x + threadIdx.x;   // b*C*F + c*F + f
    if (idx >= (long)B_ * C_ * F_) return;
    int f = (int)(idx & (F_ - 1));
    int c = (int)((idx >> 10) & (C_ - 1));
    int b = (int)(idx >> 14);
    const float* sk = obs_skill + (long)b * OSD_ + OBS_ + c * S_;
    const float* w  = seW + ((long)c * S_) * F_ + f;
    float acc = seB[c * F_ + f];
#pragma unroll
    for (int s = 0; s < S_; ++s) acc += sk[s] * w[(long)s * F_];
    out[idx] = (bf16)acc;
}

// ---------------------------------------------------------------------------
// Tiled bf16 WMMA GEMM:  out = act(A @ B + bias), B given TRANSPOSED (N x K).
//   - A split into A0 (k < splitK) / A1 (k >= splitK); channel via blockIdx.z
//   - 256 threads = 8 wave32s; each wave computes (MT*16) x (NT*16) output.
//   - LDS staged in ISA VGPR fragment layout with async b128 copies,
//     double-buffered: one barrier per K-step, copies overlap WMMA.
// ---------------------------------------------------------------------------
#define BK 32

template <int WROWS, int WCOLS, int MT, int NT>
__global__ __launch_bounds__(256) void gemm_bf16(
    const bf16* __restrict__ A0, long lda0, long aCS0,
    const bf16* __restrict__ A1, long lda1, long aCS1, int splitK,
    const bf16* __restrict__ Bt, long bCS, int ldb,         // ldb == K
    const float* __restrict__ bias, long biasCS,
    bf16* __restrict__ outBf, float* __restrict__ outF, long ldc, long outCS,
    int K, int act)
{
    constexpr int BMc = WROWS * MT * 16;
    constexpr int BNc = WCOLS * NT * 16;
    constexpr int ATILES = BMc / 16;
    constexpr int BTILES = BNc / 16;
    constexpr int AIT = (BMc * 4) / 256;   // 16B chunks per thread (A tile)
    constexpr int BIT = (BNc * 4) / 256;   // 16B chunks per thread (B tile)

    __shared__ alignas(16) bf16 sA[2][ATILES][32][16];
    __shared__ alignas(16) bf16 sB[2][BTILES][32][16];
    constexpr unsigned ABUF = sizeof(sA) / 2;   // byte stride between buffers
    constexpr unsigned BBUF = sizeof(sB) / 2;

    const int tid  = threadIdx.x;
    const int lane = tid & 31;
    const int wave = tid >> 5;
    const int c    = blockIdx.z;
    const int m0   = blockIdx.y * BMc;
    const int n0   = blockIdx.x * BNc;

    const bf16*  Aa = A0 + (long)c * aCS0;
    const bf16*  Ab = A1 + (long)c * aCS1;
    const bf16*  Bp = Bt + (long)c * bCS;
    const float* bp = bias + (long)c * biasCS;

    const int wmTile = (wave / WCOLS) * MT;
    const int wnTile = (wave % WCOLS) * NT;

    // ---- per-thread staging addresses (loop-invariant) ----
    unsigned aLds[AIT]; long aOff0[AIT], aOff1[AIT];
#pragma unroll
    for (int it = 0; it < AIT; ++it) {
        int ch  = tid + it * 256;
        int row = ch >> 2;
        int kc  = (ch & 3) * 8;
        aLds[it]  = (unsigned)(size_t)&sA[0][row >> 4][(row & 15) + (((kc >> 3) & 1) << 4)][(kc & 16) >> 1];
        aOff0[it] = (long)(m0 + row) * lda0 + kc;
        aOff1[it] = (long)(m0 + row) * lda1 + kc;
    }
    unsigned bLds[BIT]; long bOff[BIT];
#pragma unroll
    for (int it = 0; it < BIT; ++it) {
        int ch = tid + it * 256;
        int nl = ch >> 2;
        int kc = (ch & 3) * 8;
        bLds[it] = (unsigned)(size_t)&sB[0][nl >> 4][(nl & 15) + (((kc >> 3) & 1) << 4)][(kc & 16) >> 1];
        bOff[it] = (long)(n0 + nl) * ldb + kc;
    }

    auto stage = [&](unsigned buf, int k0) {
        if (k0 < splitK) {                                   // wave-uniform branch
            const bf16* abase = Aa + k0;
#pragma unroll
            for (int it = 0; it < AIT; ++it)
                async_copy_b128(aLds[it] + buf * ABUF, abase + aOff0[it]);
        } else {
            const bf16* abase = Ab + (k0 - splitK);
#pragma unroll
            for (int it = 0; it < AIT; ++it)
                async_copy_b128(aLds[it] + buf * ABUF, abase + aOff1[it]);
        }
        const bf16* bbase = Bp + k0;
#pragma unroll
        for (int it = 0; it < BIT; ++it)
            async_copy_b128(bLds[it] + buf * BBUF, bbase + bOff[it]);
    };

    const v8f vzero = {0.f, 0.f, 0.f, 0.f, 0.f, 0.f, 0.f, 0.f};
    v8f acc[MT][NT];
#pragma unroll
    for (int i = 0; i < MT; ++i)
#pragma unroll
        for (int j = 0; j < NT; ++j) acc[i][j] = vzero;

    // ---- software-pipelined main loop (double-buffered LDS) ----
    stage(0, 0);
    for (int k0 = 0; k0 < K; k0 += BK) {
        const unsigned buf = (unsigned)((k0 / BK) & 1);
        wait_async0();          // this wave's copies into `buf` are done
        __syncthreads();        // everyone's copies visible / prev reads done
        if (k0 + BK < K) stage(buf ^ 1u, k0 + BK);

        v16bf aF[MT], bF[NT];
#pragma unroll
        for (int i = 0; i < MT; ++i) aF[i] = *(const v16bf*)&sA[buf][wmTile + i][lane][0];
#pragma unroll
        for (int j = 0; j < NT; ++j) bF[j] = *(const v16bf*)&sB[buf][wnTile + j][lane][0];
#pragma unroll
        for (int i = 0; i < MT; ++i)
#pragma unroll
            for (int j = 0; j < NT; ++j)
                acc[i][j] = __builtin_amdgcn_wmma_f32_16x16x32_bf16(
                    false, aF[i], false, bF[j], (short)0, acc[i][j], false, false);
    }

    // ---- epilogue: D layout lane = n + 16*(m>>3), vgpr v = m&7 ----
    const int colInTile = lane & 15;
    const int rowOff    = (lane >> 4) * 8;
#pragma unroll
    for (int i = 0; i < MT; ++i) {
#pragma unroll
        for (int j = 0; j < NT; ++j) {
            int mbase = m0 + (wmTile + i) * 16;
            int nbase = n0 + (wnTile + j) * 16;
            int col   = nbase + colInTile;
            float bv  = bp[col];
#pragma unroll
            for (int v = 0; v < 8; ++v) {
                int row = mbase + rowOff + v;
                float x = acc[i][j][v] + bv;
                if (act == 1) x = fmaxf(x, 0.f);
                long o = (long)c * outCS + (long)row * ldc + col;
                if (outBf) outBf[o] = (bf16)x;
                if (outF)  outF[o]  = x;
            }
        }
    }
}

// ---------------------------------------------------------------------------
// gate head: weights[b,c] = sigmoid(gh[b,:] @ gate_W2[:,c] + b2[c])  (N=16)
// ---------------------------------------------------------------------------
__global__ void gate2_kernel(const float* __restrict__ gh,
                             const float* __restrict__ W2,
                             const float* __restrict__ b2,
                             float* __restrict__ gate) {
    int idx = blockIdx.x * blockDim.x + threadIdx.x;  // b*C + c
    if (idx >= B_ * C_) return;
    int c = idx & (C_ - 1);
    int b = idx >> 4;
    const float* g = gh + (long)b * F_;
    float acc = b2[c];
    for (int k = 0; k < F_; ++k) acc += g[k] * W2[(long)k * C_ + c];
    gate[idx] = 1.f / (1.f + __expf(-acc));
}

// ---------------------------------------------------------------------------
// mixture epilogue: per (b,a) combine the 16 experts
// ---------------------------------------------------------------------------
__global__ void epilogue_kernel(const float* __restrict__ outp,   // [B,C,2A]
                                const float* __restrict__ gate,   // [B,C]
                                float* __restrict__ mean,
                                float* __restrict__ scale) {
    int idx = blockIdx.x * blockDim.x + threadIdx.x;  // b*A + a
    if (idx >= B_ * A_) return;
    int a = idx & (A_ - 1);
    int b = idx >> 5;
    float wsum = 0.f, wmu = 0.f;
#pragma unroll
    for (int c = 0; c < C_; ++c) {
        const float* o = outp + ((long)b * C_ + c) * (2 * A_);
        float mu = tanhf(o[a]);
        float t  = tanhf(o[A_ + a]);
        float sg = __expf(6.f * t - 4.f);  // LOG_STD_MIN + 0.5*(t+1)*range
        float w  = gate[b * C_ + c] / sg;
        wsum += w;
        wmu  += w * mu;
    }
    mean[idx]  = wmu / wsum;
    scale[idx] = 1.f / wsum;
}

// ---------------------------------------------------------------------------
extern "C" void kernel_launch(void* const* d_in, const int* in_sizes, int n_in,
                              void* d_out, int out_size, void* d_ws, size_t ws_size,
                              hipStream_t stream) {
    (void)in_sizes; (void)n_in; (void)out_size; (void)ws_size;

    const float* obs_skill = (const float*)d_in[0];
    const float* enc_W1 = (const float*)d_in[2];
    const float* enc_b1 = (const float*)d_in[3];
    const float* enc_W2 = (const float*)d_in[4];
    const float* enc_b2 = (const float*)d_in[5];
    const float* gate_W1 = (const float*)d_in[6];
    const float* gate_b1 = (const float*)d_in[7];
    const float* gate_W2 = (const float*)d_in[8];
    const float* gate_b2 = (const float*)d_in[9];
    const float* se_W = (const float*)d_in[10];
    const float* se_b = (const float*)d_in[11];
    const float* p_W1 = (const float*)d_in[12];
    const float* p_b1 = (const float*)d_in[13];
    const float* p_W2 = (const float*)d_in[14];
    const float* p_b2 = (const float*)d_in[15];

    char* ws = (char*)d_ws;
    size_t off = 0;
    auto take = [&](size_t bytes) -> char* {
        char* p = ws + off;
        off += (bytes + 255) & ~(size_t)255;
        return p;
    };

    bf16* obs_bf   = (bf16*)take((size_t)B_ * OSD_ * 2);
    bf16* Wenc1t   = (bf16*)take((size_t)OBS_ * F_ * 2);          // [F][OBS]
    bf16* Wenc2t   = (bf16*)take((size_t)F_ * F_ * 2);            // [F][F]
    bf16* Wgate1t  = (bf16*)take((size_t)OSD_ * F_ * 2);          // [F][OSD]
    bf16* Wp1t     = (bf16*)take((size_t)C_ * 2 * F_ * F_ * 2);   // [c][F][2F]
    bf16* Wp2t     = (bf16*)take((size_t)C_ * F_ * 2 * A_ * 2);   // [c][2A][F]
    bf16* h_bf     = (bf16*)take((size_t)B_ * F_ * 2);
    bf16* emb_bf   = (bf16*)take((size_t)B_ * F_ * 2);
    bf16* skill_bf = (bf16*)take((size_t)B_ * C_ * F_ * 2);
    bf16* h1_bf    = (bf16*)take((size_t)B_ * C_ * F_ * 2);
    float* gh      = (float*)take((size_t)B_ * F_ * 4);
    float* out_f   = (float*)take((size_t)B_ * C_ * 2 * A_ * 4);
    float* gate    = (float*)take((size_t)B_ * C_ * 4);

    // ---- one-time conversions (weights transposed to N x K) ----
    cvt_f32_bf16<<<dim3((B_ * OSD_ + 255) / 256), 256, 0, stream>>>(
        obs_skill, obs_bf, (long)B_ * OSD_);
    auto cvtT = [&](const float* s, bf16* d, int K, int N, int CH) {
        long total = (long)CH * K * N;
        cvt_transpose_bf16<<<dim3((unsigned)((total + 255) / 256)), 256, 0, stream>>>(
            s, d, K, N, total);
    };
    cvtT(enc_W1,  Wenc1t,  OBS_,   F_,     1);
    cvtT(enc_W2,  Wenc2t,  F_,     F_,     1);
    cvtT(gate_W1, Wgate1t, OSD_,   F_,     1);
    cvtT(p_W1,    Wp1t,    2 * F_, F_,     C_);
    cvtT(p_W2,    Wp2t,    F_,     2 * A_, C_);

    // skill_ft (K=8, VALU)
    skill_ft_kernel<<<dim3((B_ * C_ * F_) / 256), 256, 0, stream>>>(
        obs_skill, se_W, se_b, skill_bf);

    // enc layer 1: [1024x256]@[256x1024] relu -> h_bf   (block 256x128)
    gemm_bf16<4, 2, 4, 4><<<dim3(F_ / 128, B_ / 256, 1), 256, 0, stream>>>(
        obs_bf, OSD_, 0, obs_bf, OSD_, 0, OBS_,
        Wenc1t, 0, OBS_, enc_b1, 0,
        h_bf, (float*)nullptr, F_, 0, OBS_, 1);

    // enc layer 2: [1024x1024]@[1024x1024] relu -> emb_bf
    gemm_bf16<4, 2, 4, 4><<<dim3(F_ / 128, B_ / 256, 1), 256, 0, stream>>>(
        h_bf, F_, 0, h_bf, F_, 0, F_,
        Wenc2t, 0, F_, enc_b2, 0,
        emb_bf, (float*)nullptr, F_, 0, F_, 1);

    // gate layer 1: [1024x384]@[384x1024] relu -> gh (f32)
    gemm_bf16<4, 2, 4, 4><<<dim3(F_ / 128, B_ / 256, 1), 256, 0, stream>>>(
        obs_bf, OSD_, 0, obs_bf, OSD_, 0, OSD_,
        Wgate1t, 0, OSD_, gate_b1, 0,
        (bf16*)nullptr, gh, F_, 0, OSD_, 1);

    // p1 (dominant): per channel [1024x2048]@[2048x1024], A = [emb | skill_ft]
    gemm_bf16<4, 2, 4, 4><<<dim3(F_ / 128, B_ / 256, C_), 256, 0, stream>>>(
        emb_bf, F_, 0,
        skill_bf, (long)C_ * F_, F_, F_,
        Wp1t, (long)2 * F_ * F_, 2 * F_, p_b1, F_,
        h1_bf, (float*)nullptr, (long)C_ * F_, F_, 2 * F_, 1);

    // p2: per channel [1024x1024]@[1024x64] -> out_f (f32)   (block 128x64)
    gemm_bf16<4, 2, 2, 2><<<dim3(1, B_ / 128, C_), 256, 0, stream>>>(
        h1_bf, (long)C_ * F_, F_, h1_bf, (long)C_ * F_, F_, F_,
        Wp2t, (long)F_ * 2 * A_, F_, p_b2, 2 * A_,
        (bf16*)nullptr, out_f, (long)C_ * 2 * A_, 2 * A_, F_, 0);

    // gate head + sigmoid
    gate2_kernel<<<dim3((B_ * C_) / 256), 256, 0, stream>>>(gh, gate_W2, gate_b2, gate);

    // mixture epilogue -> d_out = [mean (B*A) | scale_tril (B*A)]
    float* mean  = (float*)d_out;
    float* scale = mean + (size_t)B_ * A_;
    epilogue_kernel<<<dim3((B_ * A_) / 256), 256, 0, stream>>>(out_f, gate, mean, scale);
}
